// GroupedQueryAttention_23613730194043
// MI455X (gfx1250) — matrix operand order repaired and checked by reading, not compiled, not run on hardware
//
#include <hip/hip_runtime.h>
#include <hip/hip_bf16.h>

// ---- problem constants (match reference) ----
#define B_      4
#define T_      512
#define DIN_    4096
#define H_      32
#define G_      8
#define HD_     128
#define GS_     4
#define CACHE_  3584
#define S_      4096

typedef __attribute__((ext_vector_type(16))) __bf16    v16bf;
typedef __attribute__((ext_vector_type(8)))  float     v8f;
typedef __attribute__((ext_vector_type(4)))  unsigned  u32x4;
typedef __attribute__((ext_vector_type(8)))  int       i32x8;
typedef __attribute__((ext_vector_type(4)))  int       i32x4;

// Tensor Data Mover availability (probe-verified builtin; arity differs by
// toolchain: 5 args on ROCm7.2/clang-22, 6 args on clang-23 therock).
#if defined(__has_builtin)
#  if __has_builtin(__builtin_amdgcn_tensor_load_to_lds) && \
      __has_builtin(__builtin_amdgcn_s_wait_tensorcnt)
#    define ATH_TDM 1
#  endif
#endif
#ifndef ATH_TDM
#  define ATH_TDM 0
#endif

// ---------------------------------------------------------------------------
// WMMA fragment loaders (CDNA5 16x16x32 bf16 layouts, wave32).
// A (16x32, MxK): lanes 0-15 -> M=lane, K pairs {0..7}+{16..23};
//                 lanes 16-31 -> M=lane-16, K pairs {8..15}+{24..31}.
// B held as Bt[n][k] (k contiguous): lanes 0-15 -> N=lane, K {0..7}+{16..23};
//                 lanes 16-31 -> N=lane-16, K {8..15}+{24..31} (2 K per dword).
// ---------------------------------------------------------------------------
__device__ inline v16bf load_a_frag(const __bf16* base, int ld) {
  int lane = threadIdx.x & 31;
  int half = lane >> 4, m = lane & 15, koff = half * 8;
  union { v16bf v; unsigned u[8]; } f;
  const __bf16* row = base + m * ld;
#pragma unroll
  for (int i = 0; i < 4; ++i) f.u[i]     = *(const unsigned*)(row + koff + 2 * i);
#pragma unroll
  for (int i = 0; i < 4; ++i) f.u[4 + i] = *(const unsigned*)(row + 16 + koff + 2 * i);
  return f.v;
}

__device__ inline v16bf load_bt_frag(const __bf16* base, int ld) {
  int lane = threadIdx.x & 31;
  int half = lane >> 4, n = lane & 15, kbase = half * 8;
  union { v16bf v; unsigned u[8]; } f;
  const __bf16* row = base + n * ld;
#pragma unroll
  for (int i = 0; i < 4; ++i) f.u[i]     = *(const unsigned*)(row + kbase + 2 * i);
#pragma unroll
  for (int i = 0; i < 4; ++i) f.u[4 + i] = *(const unsigned*)(row + 16 + kbase + 2 * i);
  return f.v;
}

// ---------------------------------------------------------------------------
// Elementwise f32 -> bf16 (8 elems/thread, 16B stores)
// ---------------------------------------------------------------------------
__global__ __launch_bounds__(256) void cvt_f32_to_bf16(
    const float* __restrict__ in, __bf16* __restrict__ out) {
  size_t i = ((size_t)blockIdx.x * 256 + threadIdx.x) * 8;
  float4 a = *(const float4*)(in + i);
  float4 b = *(const float4*)(in + i + 4);
  union { __bf16 h[8]; uint4 u; } p;
  p.h[0] = (__bf16)a.x; p.h[1] = (__bf16)a.y; p.h[2] = (__bf16)a.z; p.h[3] = (__bf16)a.w;
  p.h[4] = (__bf16)b.x; p.h[5] = (__bf16)b.y; p.h[6] = (__bf16)b.z; p.h[7] = (__bf16)b.w;
  *(uint4*)(out + i) = p.u;
}

// ---------------------------------------------------------------------------
// GEMM: C[M,N] f32 = A[M,K] bf16 x W[K,N] bf16.
// Block = 256 thr (8 waves), 128(M) x 256(N) tile, 64x64 per wave (16 WMMA /
// K-step), K-step 32. Register-pipelined staging: next K-tile's global loads
// are issued during the current tile's WMMA compute.
// Requires M%128==0, N%256==0, K%32==0.
// ---------------------------------------------------------------------------
__global__ __launch_bounds__(256) void gemm_bf16_wmma(
    const __bf16* __restrict__ A, const __bf16* __restrict__ W,
    float* __restrict__ C, int M, int N, int K) {
  __shared__ __align__(16) __bf16 As[128][32];   // [m][k]
  __shared__ __align__(16) __bf16 Bs[256][32];   // Bt: [n][k]

  const int tid  = threadIdx.x;
  const int lane = tid & 31;
  const int w    = tid >> 5;
  const int wrow = w >> 2;          // 0..1 (64 rows)
  const int wcol = w & 3;           // 0..3 (64 cols)
  const int m0 = blockIdx.y * 128;
  const int n0 = blockIdx.x * 256;

  // staging coordinates
  const int ar  = tid >> 2,  ac8 = (tid & 3) * 8;    // A rows ar, ar+64
  const int wc  = tid >> 5,  wn8 = (tid & 31) * 8;   // W rows wc+8q

  v8f acc[4][4];
#pragma unroll
  for (int i = 0; i < 4; ++i)
#pragma unroll
    for (int j = 0; j < 4; ++j) acc[i][j] = (v8f){};

  uint4 rA0, rA1, rW[4];
  auto load_tile = [&](int k0) {
    rA0 = *(const uint4*)&A[(size_t)(m0 + ar) * K + k0 + ac8];
    rA1 = *(const uint4*)&A[(size_t)(m0 + ar + 64) * K + k0 + ac8];
#pragma unroll
    for (int q = 0; q < 4; ++q)
      rW[q] = *(const uint4*)&W[(size_t)(k0 + wc + 8 * q) * N + n0 + wn8];
  };
  auto store_tile = [&]() {
    *(uint4*)&As[ar][ac8]      = rA0;
    *(uint4*)&As[ar + 64][ac8] = rA1;
#pragma unroll
    for (int q = 0; q < 4; ++q) {
      union { uint4 u; __bf16 h[8]; } p;
      p.u = rW[q];
#pragma unroll
      for (int e = 0; e < 8; ++e) Bs[wn8 + e][wc + 8 * q] = p.h[e];
    }
  };

  load_tile(0);
  for (int k0 = 0; k0 < K; k0 += 32) {
    __syncthreads();               // previous tile's readers are done
    store_tile();
    __syncthreads();
    if (k0 + 32 < K) load_tile(k0 + 32);   // overlap with WMMA below

    v16bf bf[4];
#pragma unroll
    for (int mt = 0; mt < 4; ++mt) {
      v16bf af = load_a_frag(&As[wrow * 64 + mt * 16][0], 32);
#pragma unroll
      for (int nt = 0; nt < 4; ++nt) {
        if (mt == 0) bf[nt] = load_bt_frag(&Bs[wcol * 64 + nt * 16][0], 32);
        acc[mt][nt] = __builtin_amdgcn_wmma_f32_16x16x32_bf16(
            false, af, false, bf[nt], (short)0, acc[mt][nt], false, false);
      }
    }
  }

  const int half = lane >> 4, nn = lane & 15;
#pragma unroll
  for (int mt = 0; mt < 4; ++mt)
#pragma unroll
    for (int nt = 0; nt < 4; ++nt)
#pragma unroll
      for (int j = 0; j < 8; ++j) {
        int r = m0 + wrow * 64 + mt * 16 + j + 8 * half;
        int c = n0 + wcol * 64 + nt * 16 + nn;
        C[(size_t)r * N + c] = acc[mt][nt][j];
      }
}

// ---------------------------------------------------------------------------
// RoPE on Q: read f32 ws, write bf16 [B,T,H,HD], pos = CACHE + t
// ---------------------------------------------------------------------------
__global__ __launch_bounds__(256) void rope_q_kernel(
    const float* __restrict__ q, __bf16* __restrict__ qb,
    const float* __restrict__ cosT, const float* __restrict__ sinT) {
  int i = blockIdx.x * 256 + threadIdx.x;          // B*T*H*64 threads
  int hd2 = i & 63;
  int tmp = i >> 6;
  int h = tmp & 31;  tmp >>= 5;
  int t = tmp & 511; int b = tmp >> 9;
  int pos = CACHE_ + t;
  size_t base = (((size_t)(b * T_ + t)) * H_ + h) * HD_;
  float c1 = cosT[(size_t)pos * HD_ + hd2];
  float s1 = sinT[(size_t)pos * HD_ + hd2];
  float c2 = cosT[(size_t)pos * HD_ + hd2 + 64];
  float s2 = sinT[(size_t)pos * HD_ + hd2 + 64];
  float x1 = q[base + hd2];
  float x2 = q[base + hd2 + 64];
  qb[base + hd2]      = (__bf16)(x1 * c1 - x2 * s1);
  qb[base + hd2 + 64] = (__bf16)(x2 * c2 + x1 * s2);
}

// ---------------------------------------------------------------------------
// KV cache assembly: f32 k_all/v_all outputs [B,G,S,HD] + bf16 mirrors.
// New K gets RoPE at pos = s. kws/vws: [B,T,G,HD] f32.
// ---------------------------------------------------------------------------
__global__ __launch_bounds__(256) void kv_cache_kernel(
    const float* __restrict__ kws, const float* __restrict__ vws,
    const float* __restrict__ prev_k, const float* __restrict__ prev_v,
    const float* __restrict__ cosT, const float* __restrict__ sinT,
    float* __restrict__ k_all, float* __restrict__ v_all,
    __bf16* __restrict__ kb, __bf16* __restrict__ vb) {
  int i = blockIdx.x * 256 + threadIdx.x;          // B*G*S*64 threads
  int hd2 = i & 63;
  int tmp = i >> 6;
  int s = tmp & (S_ - 1); tmp >>= 12;
  int g = tmp & (G_ - 1); int b = tmp >> 3;
  size_t obase = (((size_t)(b * G_ + g)) * S_ + s) * HD_;
  float k1, k2, w1, w2;
  if (s < CACHE_) {
    size_t pbase = (((size_t)(b * G_ + g)) * CACHE_ + s) * HD_;
    k1 = prev_k[pbase + hd2];  k2 = prev_k[pbase + hd2 + 64];
    w1 = prev_v[pbase + hd2];  w2 = prev_v[pbase + hd2 + 64];
  } else {
    int t = s - CACHE_;
    size_t ibase = (((size_t)(b * T_ + t)) * G_ + g) * HD_;
    float c1 = cosT[(size_t)s * HD_ + hd2];
    float s1 = sinT[(size_t)s * HD_ + hd2];
    float c2 = cosT[(size_t)s * HD_ + hd2 + 64];
    float s2 = sinT[(size_t)s * HD_ + hd2 + 64];
    float x1 = kws[ibase + hd2];
    float x2 = kws[ibase + hd2 + 64];
    k1 = x1 * c1 - x2 * s1;
    k2 = x2 * c2 + x1 * s2;
    w1 = vws[ibase + hd2];
    w2 = vws[ibase + hd2 + 64];
  }
  k_all[obase + hd2]      = k1;  k_all[obase + hd2 + 64] = k2;
  v_all[obase + hd2]      = w1;  v_all[obase + hd2 + 64] = w2;
  kb[obase + hd2]      = (__bf16)k1;  kb[obase + hd2 + 64] = (__bf16)k2;
  vb[obase + hd2]      = (__bf16)w1;  vb[obase + hd2 + 64] = (__bf16)w2;
}

// ---------------------------------------------------------------------------
// Flash-style GQA attention (bf16 operands, f32 accum/softmax).
// Block = (128-query tile, b*h), 256 threads; wave w owns rows [16w,16w+16).
// K chunk staged by the Tensor Data Mover; V chunk register-pipelined.
// ---------------------------------------------------------------------------
__global__ __launch_bounds__(256) void gqa_attn_kernel(
    const __bf16* __restrict__ qb, const __bf16* __restrict__ kb,
    const __bf16* __restrict__ vb, __bf16* __restrict__ ctxb) {
  __shared__ __align__(16) __bf16 Ks[64][128];    // [s][hd] == Bt for Q*K^T
  __shared__ __align__(16) __bf16 Vts[128][64];   // [hd][s] == Bt for P*V
  __shared__ __align__(16) __bf16 Ps[128][64];    // P staging

  const int tid  = threadIdx.x;
  const int lane = tid & 31;
  const int half = lane >> 4;
  const int nn   = lane & 15;
  const int w    = tid >> 5;
  const int t0 = blockIdx.x * 128;
  const int bh = blockIdx.y;
  const int b = bh >> 5, h = bh & 31, g = h >> 2;

  // this wave's Q fragments (rows 16w..16w+15, 4 K-steps of 32)
  const __bf16* qbase = qb + (((size_t)(b * T_ + t0 + w * 16)) * H_ + h) * HD_;
  v16bf aq[4];
#pragma unroll
  for (int kk = 0; kk < 4; ++kk)
    aq[kk] = load_a_frag(qbase + kk * 32, H_ * HD_);

  v8f o[8];
#pragma unroll
  for (int i = 0; i < 8; ++i) o[i] = (v8f){};
  float m[8], l[8];
#pragma unroll
  for (int j = 0; j < 8; ++j) { m[j] = -INFINITY; l[j] = 0.f; }

  const float scale = 0.08838834764831845f;   // 1/sqrt(128)
  const int s_end = CACHE_ + t0 + 128;        // causal frontier of this tile
  const __bf16* kbase = kb + ((size_t)(b * G_ + g)) * S_ * HD_;
  const __bf16* vbase = vb + ((size_t)(b * G_ + g)) * S_ * HD_;
#if ATH_TDM
  const unsigned ks_lds = (unsigned)(size_t)&Ks[0][0];
#endif

  // V chunk register pipeline (and K when no TDM)
  const int vr = tid >> 4, vh8 = (tid & 15) * 8;   // rows vr + 16q
  uint4 rV[4];
#if !ATH_TDM
  uint4 rK[4];
#endif
  auto load_chunk = [&](int s0) {
#pragma unroll
    for (int q = 0; q < 4; ++q)
      rV[q] = *(const uint4*)&vbase[(size_t)(s0 + vr + 16 * q) * HD_ + vh8];
#if !ATH_TDM
#pragma unroll
    for (int q = 0; q < 4; ++q)
      rK[q] = *(const uint4*)&kbase[(size_t)(s0 + vr + 16 * q) * HD_ + vh8];
#endif
  };

  load_chunk(0);
  for (int s0 = 0; s0 < s_end; s0 += 64) {
    __syncthreads();               // previous chunk's readers are done
#if ATH_TDM
    // ---- K chunk via Tensor Data Mover: 128(x) x 64(y) bf16 tile ----
    if (w == 0) {
      unsigned long long ga =
          (unsigned long long)(size_t)(kbase + (size_t)s0 * HD_);
      u32x4 g0;
      g0[0] = 1u;                                   // count=1 (user D#)
      g0[1] = ks_lds;                               // lds_addr
      g0[2] = (unsigned)ga;                         // global_addr[31:0]
      g0[3] = (unsigned)((ga >> 32) & 0x1FFFFFFull) // global_addr[56:32]
              | (2u << 30);                         // type=2 ("image")
      i32x8 g1;
      g1[0] = (int)(1u << 16);              // data_size = 2 bytes
      g1[1] = (int)((unsigned)HD_ << 16);   // tensor_dim0 = 128
      g1[2] = (int)((unsigned)S_ << 16);    // tensor_dim1 = 4096
      g1[3] = (int)((unsigned)HD_ << 16);   // tile_dim0 = 128
      g1[4] = 64;                           // tile_dim1 = 64
      g1[5] = HD_;                          // tensor_dim0_stride = 128
      g1[6] = 0;
      g1[7] = 0;
      i32x4 gz = (i32x4){0, 0, 0, 0};
#if __clang_major__ >= 23
      i32x8 gz8 = (i32x8){0, 0, 0, 0, 0, 0, 0, 0};
      __builtin_amdgcn_tensor_load_to_lds(g0, g1, gz, gz, gz8, 0);
#else
      __builtin_amdgcn_tensor_load_to_lds(g0, g1, gz, gz, 0);
#endif
    }
#else
    // ---- K chunk from prefetched registers ----
#pragma unroll
    for (int q = 0; q < 4; ++q)
      *(uint4*)&Ks[vr + 16 * q][vh8] = rK[q];
#endif
    // ---- V chunk transpose from prefetched registers ----
#pragma unroll
    for (int q = 0; q < 4; ++q) {
      union { uint4 u; __bf16 hh[8]; } p;
      p.u = rV[q];
#pragma unroll
      for (int e = 0; e < 8; ++e) Vts[vh8 + e][vr + 16 * q] = p.hh[e];
    }
#if ATH_TDM
    __builtin_amdgcn_s_wait_tensorcnt(0);
#endif
    __syncthreads();
    if (s0 + 64 < s_end) load_chunk(s0 + 64);   // overlap with compute below

    // ---- scores: S = Q x K^T (16 x 64 per wave) ----
    v8f sc[4];
#pragma unroll
    for (int nt = 0; nt < 4; ++nt) sc[nt] = (v8f){};
#pragma unroll
    for (int kk = 0; kk < 4; ++kk) {
#pragma unroll
      for (int nt = 0; nt < 4; ++nt) {
        v16bf bk = load_bt_frag(&Ks[nt * 16][kk * 32], 128);
        sc[nt] = __builtin_amdgcn_wmma_f32_16x16x32_bf16(
            false, aq[kk], false, bk, (short)0, sc[nt], false, false);
      }
    }

    // ---- scale + causal mask + chunk row-max ----
    float cm[8];
#pragma unroll
    for (int j = 0; j < 8; ++j) cm[j] = -INFINITY;
#pragma unroll
    for (int nt = 0; nt < 4; ++nt)
#pragma unroll
      for (int j = 0; j < 8; ++j) {
        float sv = sc[nt][j] * scale;
        int s_abs = s0 + nt * 16 + nn;
        int q_abs = CACHE_ + t0 + w * 16 + j + 8 * half;
        if (s_abs > q_abs) sv = -INFINITY;
        sc[nt][j] = sv;
        cm[j] = fmaxf(cm[j], sv);
      }
#pragma unroll
    for (int j = 0; j < 8; ++j)
#pragma unroll
      for (int d = 1; d < 16; d <<= 1)
        cm[j] = fmaxf(cm[j], __shfl_xor(cm[j], d, 32));

    // ---- online softmax ----
    float fac[8], mn[8], rs[8];
#pragma unroll
    for (int j = 0; j < 8; ++j) {
      mn[j]  = fmaxf(m[j], cm[j]);
      fac[j] = __expf(m[j] - mn[j]);
      m[j]   = mn[j];
      rs[j]  = 0.f;
    }
#pragma unroll
    for (int nt = 0; nt < 4; ++nt)
#pragma unroll
      for (int j = 0; j < 8; ++j) {
        float p = __expf(sc[nt][j] - mn[j]);
        sc[nt][j] = p;
        rs[j] += p;
      }
#pragma unroll
    for (int j = 0; j < 8; ++j)
#pragma unroll
      for (int d = 1; d < 16; d <<= 1)
        rs[j] += __shfl_xor(rs[j], d, 32);
#pragma unroll
    for (int j = 0; j < 8; ++j) l[j] = l[j] * fac[j] + rs[j];
#pragma unroll
    for (int hdt = 0; hdt < 8; ++hdt)
#pragma unroll
      for (int j = 0; j < 8; ++j) o[hdt][j] *= fac[j];

    // ---- stage P (bf16); same-wave LDS RAW is in-order ----
#pragma unroll
    for (int nt = 0; nt < 4; ++nt)
#pragma unroll
      for (int j = 0; j < 8; ++j)
        Ps[w * 16 + j + 8 * half][nt * 16 + nn] = (__bf16)sc[nt][j];

    // ---- O += P x V ----
#pragma unroll
    for (int kk = 0; kk < 2; ++kk) {
      v16bf ap = load_a_frag(&Ps[w * 16][kk * 32], 64);
#pragma unroll
      for (int hdt = 0; hdt < 8; ++hdt) {
        v16bf bv = load_bt_frag(&Vts[hdt * 16][kk * 32], 64);
        o[hdt] = __builtin_amdgcn_wmma_f32_16x16x32_bf16(
            false, ap, false, bv, (short)0, o[hdt], false, false);
      }
    }
  }

  // ---- normalize + write bf16 ctx [B,T,H*HD] ----
#pragma unroll
  for (int hdt = 0; hdt < 8; ++hdt)
#pragma unroll
    for (int j = 0; j < 8; ++j) {
      int t = t0 + w * 16 + j + 8 * half;
      ctxb[((size_t)(b * T_ + t)) * (H_ * HD_) + h * HD_ + hdt * 16 + nn] =
          (__bf16)(o[hdt][j] * (1.f / l[j]));
    }
}

// ---------------------------------------------------------------------------
extern "C" void kernel_launch(void* const* d_in, const int* in_sizes, int n_in,
                              void* d_out, int out_size, void* d_ws,
                              size_t ws_size, hipStream_t stream) {
  (void)in_sizes; (void)n_in; (void)out_size; (void)ws_size;
  const float* x      = (const float*)d_in[0];
  // d_in[1] = mask (bool): causal mask computed analytically
  const float* cosT   = (const float*)d_in[2];
  const float* sinT   = (const float*)d_in[3];
  // d_in[4] = start_positions: uniform == CACHE_
  const float* prev_k = (const float*)d_in[5];
  const float* prev_v = (const float*)d_in[6];
  const float* q_w    = (const float*)d_in[7];
  const float* k_w    = (const float*)d_in[8];
  const float* v_w    = (const float*)d_in[9];
  const float* o_w    = (const float*)d_in[10];

  float* out   = (float*)d_out;                               // [B,T,H*HD]
  float* k_all = out + (size_t)B_ * T_ * H_ * HD_;            // [B,G,S,HD]
  float* v_all = k_all + (size_t)B_ * G_ * S_ * HD_;

  // ---- workspace carve-up ----
  char* p = (char*)d_ws;
  auto carve = [&](size_t bytes) { char* r = p; p += (bytes + 255) & ~(size_t)255; return r; };
  const size_t NX   = (size_t)B_ * T_ * DIN_;       // 8,388,608
  const size_t NQW  = (size_t)DIN_ * H_ * HD_;      // 16,777,216
  const size_t NKW  = (size_t)DIN_ * G_ * HD_;      //  4,194,304
  const size_t NQ   = (size_t)B_ * T_ * H_ * HD_;   //  8,388,608
  const size_t NKV  = (size_t)B_ * T_ * G_ * HD_;   //  2,097,152
  const size_t NKA  = (size_t)B_ * G_ * S_ * HD_;   // 16,777,216

  __bf16* xb   = (__bf16*)carve(NX * 2);
  __bf16* qwb  = (__bf16*)carve(NQW * 2);
  __bf16* kwb  = (__bf16*)carve(NKW * 2);
  __bf16* vwb  = (__bf16*)carve(NKW * 2);
  __bf16* owb  = (__bf16*)carve(NQW * 2);
  float*  qws  = (float*)carve(NQ * 4);
  float*  kws  = (float*)carve(NKV * 4);
  float*  vws  = (float*)carve(NKV * 4);
  __bf16* qbb  = (__bf16*)carve(NQ * 2);
  __bf16* kbb  = (__bf16*)carve(NKA * 2);
  __bf16* vbb  = (__bf16*)carve(NKA * 2);
  __bf16* ctxb = (__bf16*)carve(NQ * 2);

  auto cvt = [&](const float* src, __bf16* dst, size_t n) {
    cvt_f32_to_bf16<<<(unsigned)(n / (8 * 256)), 256, 0, stream>>>(src, dst);
  };

  // 0: one-shot bf16 conversions of x and weights
  cvt(x, xb, NX);
  cvt(q_w, qwb, NQW);
  cvt(k_w, kwb, NKW);
  cvt(v_w, vwb, NKW);
  cvt(o_w, owb, NQW);

  const int M = B_ * T_;   // 2048 token rows

  // 1-3: QKV projections (bf16 WMMA, f32 out)
  gemm_bf16_wmma<<<dim3((H_ * HD_) / 256, M / 128), 256, 0, stream>>>(
      xb, qwb, qws, M, H_ * HD_, DIN_);
  gemm_bf16_wmma<<<dim3((G_ * HD_) / 256, M / 128), 256, 0, stream>>>(
      xb, kwb, kws, M, G_ * HD_, DIN_);
  gemm_bf16_wmma<<<dim3((G_ * HD_) / 256, M / 128), 256, 0, stream>>>(
      xb, vwb, vws, M, G_ * HD_, DIN_);

  // 4: RoPE on Q -> bf16
  rope_q_kernel<<<(B_ * T_ * H_ * (HD_ / 2)) / 256, 256, 0, stream>>>(
      qws, qbb, cosT, sinT);

  // 5: KV cache assembly (f32 outputs + bf16 mirrors)
  kv_cache_kernel<<<(B_ * G_ * S_ * (HD_ / 2)) / 256, 256, 0, stream>>>(
      kws, vws, prev_k, prev_v, cosT, sinT, k_all, v_all, kbb, vbb);

  // 6: attention (TDM-staged K chunks, register-pipelined V)
  gqa_attn_kernel<<<dim3(T_ / 128, B_ * H_), 256, 0, stream>>>(
      qbb, kbb, vbb, ctxb);

  // 7: output projection
  gemm_bf16_wmma<<<dim3((H_ * HD_) / 256, M / 128), 256, 0, stream>>>(
      ctxb, owb, out, M, H_ * HD_, DIN_);
}
